// ModelConcept2_43731357008307
// MI455X (gfx1250) — matrix operand order repaired and checked by reading, not compiled, not run on hardware
//
#include <hip/hip_runtime.h>
#include <hip/hip_bf16.h>

// ---------------------------------------------------------------------------
// Model dims (match reference)
// ---------------------------------------------------------------------------
static constexpr int S  = 2048;
static constexpr int E  = 1024;
static constexpr int F  = 4096;
static constexpr int H  = 8;
static constexpr int D  = 128;
static constexpr int NL = 4;
static constexpr int V  = 32000;
static constexpr int HD = H * D;   // 1024
static constexpr float NEG = -1.0e9f;

typedef __bf16 bf16;
typedef __bf16 v16bf __attribute__((ext_vector_type(16)));
typedef float  v8f   __attribute__((ext_vector_type(8)));
typedef unsigned int v4u __attribute__((ext_vector_type(4)));
typedef int v4i __attribute__((ext_vector_type(4)));

union Frag {            // 16 bf16 = 32 bytes = one WMMA A/B operand (wave32)
    v16bf v;
    v4u   u[2];
    bf16  e[16];
};

#define GLOBAL_AS __attribute__((address_space(1)))
#define LDS_AS    __attribute__((address_space(3)))

typedef GLOBAL_AS v4i* gvec_ptr;
typedef LDS_AS    v4i* lvec_ptr;

// Async 16B global -> LDS copy (per-lane), tracked by ASYNCcnt.
// Generic-pointer low 32 bits == LDS offset (ISA aperture rule), so the
// int->addrspace(3) reinterpret is well-defined; global generic == addrspace(1).
__device__ __forceinline__ void async_cp16(const void* g, void* l) {
    __builtin_amdgcn_global_load_async_to_lds_b128(
        (gvec_ptr)(uintptr_t)g,
        (lvec_ptr)(uintptr_t)l,
        0, 0);
}

// ---------------------------------------------------------------------------
// Embedding gather + fp32 -> bf16 cast:  h_bf[s][e] = embedding[x[s]][e]
// ---------------------------------------------------------------------------
__global__ __launch_bounds__(256) void gather_embed(const int* __restrict__ x,
                                                    const float* __restrict__ emb,
                                                    bf16* __restrict__ h) {
    const int s   = blockIdx.x;
    const int row = x[s];
    const float* src = emb + (size_t)row * E;
    bf16* dst = h + (size_t)s * E;
    for (int e = threadIdx.x; e < E; e += 256)
        dst[e] = (bf16)src[e];
}

// ---------------------------------------------------------------------------
// GEMM: C[M,N] = act(A[M,K] (bf16) * B (fp32->bf16 on the fly))
//   BT = false : B is [K,N] row-major (normal weights)
//   BT = true  : B is [N,K] row-major (tied-embedding logits: C = A * B^T)
// Block tile 128x128, 256 threads = 8 wave32's, each wave 4x2 WMMA subtiles.
// A tile: double-buffered LDS filled by GLOBAL_LOAD_ASYNC_TO_LDS_B128.
// B tile: fp32 prefetched to registers, converted+stored to LDS transposed.
// ---------------------------------------------------------------------------
template <bool RELU, bool BT, typename OUT_T>
__global__ __launch_bounds__(256) void gemm_bf16(const bf16* __restrict__ A,
                                                 const float* __restrict__ B,
                                                 OUT_T* __restrict__ C,
                                                 int M, int N, int K) {
    __shared__ bf16 Alds[2][128][40]; // [buf][m][k], row stride 80B (16B-aligned)
    __shared__ bf16 Blds[128][40];    // [n][k] (B stored transposed in LDS)

    const int tid  = threadIdx.x;
    const int lane = tid & 31;
    const int wv   = tid >> 5;        // 0..7
    const int wm   = wv >> 2;         // 0..1 -> 64 rows each
    const int wn   = wv & 3;          // 0..3 -> 32 cols each
    const int ln   = lane & 15;
    const int lh   = lane >> 4;

    const int gm0 = blockIdx.y * 128;
    const int gn0 = blockIdx.x * 128;

    // This thread's two A-tile chunks (row, 8-col offset) and B-tile chunks.
    const int ar0 = tid >> 1,         ac0 = (tid & 1) * 8;        // chunk c = tid
    const int ar1 = (tid + 256) >> 1, ac1 = ((tid + 256) & 1) * 8;

    v8f acc[4][2];
    const v8f vzero = {};
    #pragma unroll
    for (int i = 0; i < 4; ++i)
        #pragma unroll
        for (int j = 0; j < 2; ++j) acc[i][j] = vzero;

    const int T = K >> 5;   // K/32 tiles

    // ---- prologue: async A(0), B(0) -> regs ----
    async_cp16(&A[(size_t)(gm0 + ar0) * K + ac0], &Alds[0][ar0][ac0]);
    async_cp16(&A[(size_t)(gm0 + ar1) * K + ac1], &Alds[0][ar1][ac1]);

    float4 gB[4];
    #pragma unroll
    for (int q = 0; q < 4; ++q) {
        const int c = tid + q * 256;
        if (!BT) {
            const int k  = c >> 5;
            const int n4 = (c & 31) * 4;
            gB[q] = *(const float4*)&B[(size_t)(0 + k) * N + gn0 + n4];
        } else {
            const int n  = c >> 3;
            const int k4 = (c & 7) * 4;
            gB[q] = *(const float4*)&B[(size_t)(gn0 + n) * K + 0 + k4];
        }
    }

    for (int t = 0; t < T; ++t) {
        const int cur = t & 1;
        const int k0  = t * 32;

        // ---- convert + store this tile's B regs into LDS (transposed) ----
        #pragma unroll
        for (int q = 0; q < 4; ++q) {
            const int c = tid + q * 256;
            if (!BT) {
                const int k  = c >> 5;
                const int n4 = (c & 31) * 4;
                Blds[n4 + 0][k] = (bf16)gB[q].x;
                Blds[n4 + 1][k] = (bf16)gB[q].y;
                Blds[n4 + 2][k] = (bf16)gB[q].z;
                Blds[n4 + 3][k] = (bf16)gB[q].w;
            } else {
                const int n  = c >> 3;
                const int k4 = (c & 7) * 4;
                union { bf16 h[4]; uint2 u; } pk;
                pk.h[0] = (bf16)gB[q].x;
                pk.h[1] = (bf16)gB[q].y;
                pk.h[2] = (bf16)gB[q].z;
                pk.h[3] = (bf16)gB[q].w;
                *(uint2*)&Blds[n][k4] = pk.u;
            }
        }

        // ---- kick off next tile: async A(t+1), B(t+1) -> regs ----
        if (t + 1 < T) {
            const int nk0 = k0 + 32;
            async_cp16(&A[(size_t)(gm0 + ar0) * K + nk0 + ac0],
                       &Alds[cur ^ 1][ar0][ac0]);
            async_cp16(&A[(size_t)(gm0 + ar1) * K + nk0 + ac1],
                       &Alds[cur ^ 1][ar1][ac1]);
            #pragma unroll
            for (int q = 0; q < 4; ++q) {
                const int c = tid + q * 256;
                if (!BT) {
                    const int k  = c >> 5;
                    const int n4 = (c & 31) * 4;
                    gB[q] = *(const float4*)&B[(size_t)(nk0 + k) * N + gn0 + n4];
                } else {
                    const int n  = c >> 3;
                    const int k4 = (c & 7) * 4;
                    gB[q] = *(const float4*)&B[(size_t)(gn0 + n) * K + nk0 + k4];
                }
            }
            // A(t) complete when <= 2 outstanding (the 2 ops for t+1;
            // async loads complete in order).
            asm volatile("s_wait_asynccnt 0x2" ::: "memory");
        } else {
            asm volatile("s_wait_asynccnt 0x0" ::: "memory");
        }
        __syncthreads();   // A(t) in LDS (all waves), B(t) stores visible

        // ---- fragments + 8 WMMAs ----
        Frag af[4], bfr[2];
        #pragma unroll
        for (int i = 0; i < 4; ++i) {
            const bf16* base = &Alds[cur][wm * 64 + i * 16 + ln][0];
            af[i].u[0] = *(const v4u*)(base + lh * 8);        // K = lh*8 .. +7
            af[i].u[1] = *(const v4u*)(base + 16 + lh * 8);   // K = 16+lh*8 .. +7
        }
        #pragma unroll
        for (int j = 0; j < 2; ++j) {
            const bf16* base = &Blds[wn * 32 + j * 16 + ln][0];
            bfr[j].u[0] = *(const v4u*)(base + lh * 8);       // K = lh*8 .. +7
            bfr[j].u[1] = *(const v4u*)(base + 16 + lh * 8);  // K = 16+lh*8 .. +7
        }
        #pragma unroll
        for (int i = 0; i < 4; ++i)
            #pragma unroll
            for (int j = 0; j < 2; ++j)
                acc[i][j] = __builtin_amdgcn_wmma_f32_16x16x32_bf16(
                    false, af[i].v, false, bfr[j].v, (short)0, acc[i][j],
                    false, false);
        __syncthreads();   // free Blds and Alds[cur^1] for next iteration
    }

    // ---- epilogue: C/D layout (VGPR r, lane halves: M=r / M=r+8) ----
    #pragma unroll
    for (int i = 0; i < 4; ++i)
        #pragma unroll
        for (int j = 0; j < 2; ++j)
            #pragma unroll
            for (int r = 0; r < 8; ++r) {
                const int m = gm0 + wm * 64 + i * 16 + r + 8 * lh;
                const int n = gn0 + wn * 32 + j * 16 + ln;
                float v = acc[i][j][r];
                if (RELU) v = fmaxf(v, 0.0f);
                C[(size_t)m * N + n] = (OUT_T)v;
            }
}

// ---------------------------------------------------------------------------
// Head reorders (bf16):  [S, H*D] -> [H][S][D]  and  [S, H*D] -> [H][D][S]
// ---------------------------------------------------------------------------
__global__ __launch_bounds__(256) void reorder_hsd(const bf16* __restrict__ src,
                                                   bf16* __restrict__ dst) {
    const int s = blockIdx.x;
    for (int hd = threadIdx.x; hd < HD; hd += 256)
        dst[(size_t)(hd >> 7) * S * D + (size_t)s * D + (hd & 127)] =
            src[(size_t)s * HD + hd];
}
__global__ __launch_bounds__(256) void reorder_hds(const bf16* __restrict__ src,
                                                   bf16* __restrict__ dst) {
    const int s = blockIdx.x;
    for (int hd = threadIdx.x; hd < HD; hd += 256)
        dst[(size_t)(hd >> 7) * D * S + (size_t)(hd & 127) * S + s] =
            src[(size_t)s * HD + hd];
}

// ---------------------------------------------------------------------------
// Flash attention (causal, no scale — matches reference).
// grid = (S/64, H); 128 threads = 4 waves; each wave owns 16 query rows.
// Qh,Kh: [H][S][D] bf16;  Vt: [H][D][S] bf16;  O: [S][H*D] bf16.
// ---------------------------------------------------------------------------
__global__ __launch_bounds__(128) void flash_attn(const bf16* __restrict__ Qh,
                                                  const bf16* __restrict__ Kh,
                                                  const bf16* __restrict__ Vt,
                                                  bf16* __restrict__ O) {
    __shared__ bf16 Plds[4][16][40];   // per-wave P tile, 16B-aligned rows

    const int tid  = threadIdx.x;
    const int lane = tid & 31;
    const int wv   = tid >> 5;
    const int ln   = lane & 15;
    const int lh   = lane >> 4;

    const int h   = blockIdx.y;
    const int qb  = blockIdx.x * 64 + wv * 16;   // this wave's first query row

    const bf16* Qb = Qh + (size_t)h * S * D;
    const bf16* Kb = Kh + (size_t)h * S * D;
    const bf16* Vb = Vt + (size_t)h * D * S;

    // Q fragments (A layout), kept in registers for the whole kernel.
    Frag qf[4];
    {
        const bf16* qrow = Qb + (size_t)(qb + ln) * D;
        #pragma unroll
        for (int f = 0; f < 4; ++f) {
            qf[f].u[0] = *(const v4u*)(qrow + f * 32 + lh * 8);
            qf[f].u[1] = *(const v4u*)(qrow + f * 32 + 16 + lh * 8);
        }
    }

    const v8f vzero = {};
    v8f o_acc[8];
    #pragma unroll
    for (int t = 0; t < 8; ++t) o_acc[t] = vzero;
    float mrow[8], lrow[8];
    #pragma unroll
    for (int r = 0; r < 8; ++r) { mrow[r] = -1.0e30f; lrow[r] = 0.0f; }

    for (int kb = 0; kb < qb + 16; kb += 32) {
        // ---- scores: two 16x16 tiles over 32 keys, contracting D=128 ----
        v8f s0 = vzero, s1 = vzero;
        #pragma unroll
        for (int f = 0; f < 4; ++f) {
            Frag kf0, kf1;
            const bf16* k0p = Kb + (size_t)(kb + ln) * D + f * 32 + lh * 8;
            const bf16* k1p = Kb + (size_t)(kb + 16 + ln) * D + f * 32 + lh * 8;
            kf0.u[0] = *(const v4u*)(k0p);     kf0.u[1] = *(const v4u*)(k0p + 16);
            kf1.u[0] = *(const v4u*)(k1p);     kf1.u[1] = *(const v4u*)(k1p + 16);
            s0 = __builtin_amdgcn_wmma_f32_16x16x32_bf16(false, qf[f].v, false,
                    kf0.v, (short)0, s0, false, false);
            s1 = __builtin_amdgcn_wmma_f32_16x16x32_bf16(false, qf[f].v, false,
                    kf1.v, (short)0, s1, false, false);
        }

        // ---- causal mask + online softmax (rows duplicated per 16-lane half)
        #pragma unroll
        for (int r = 0; r < 8; ++r) {
            const int qi  = qb + r + 8 * lh;
            const int kj0 = kb + ln;
            const int kj1 = kb + 16 + ln;
            float a = (kj0 <= qi) ? s0[r] : NEG;
            float b = (kj1 <= qi) ? s1[r] : NEG;

            float mx = fmaxf(a, b);
            #pragma unroll
            for (int off = 1; off < 16; off <<= 1)
                mx = fmaxf(mx, __shfl_xor(mx, off, 16));

            const float mnew  = fmaxf(mrow[r], mx);
            const float scale = __expf(mrow[r] - mnew);
            const float p0 = __expf(a - mnew);
            const float p1 = __expf(b - mnew);
            float rs = p0 + p1;
            #pragma unroll
            for (int off = 1; off < 16; off <<= 1)
                rs += __shfl_xor(rs, off, 16);

            lrow[r] = lrow[r] * scale + rs;
            mrow[r] = mnew;
            #pragma unroll
            for (int t = 0; t < 8; ++t) o_acc[t][r] *= scale;

            const int m = r + 8 * lh;
            Plds[wv][m][ln]      = (bf16)p0;
            Plds[wv][m][16 + ln] = (bf16)p1;
        }
        // within-wave LDS RAW: wait for the P stores before the frag reads
        asm volatile("s_wait_dscnt 0x0" ::: "memory");

        // ---- P (A-frag) x V (B-frags from [D][S]) ----
        Frag pf;
        {
            const bf16* base = &Plds[wv][ln][0];
            pf.u[0] = *(const v4u*)(base + lh * 8);
            pf.u[1] = *(const v4u*)(base + 16 + lh * 8);
        }
        #pragma unroll
        for (int t = 0; t < 8; ++t) {
            Frag vf;
            const bf16* vp = Vb + (size_t)(t * 16 + ln) * S + kb + lh * 8;
            vf.u[0] = *(const v4u*)(vp);
            vf.u[1] = *(const v4u*)(vp + 16);
            o_acc[t] = __builtin_amdgcn_wmma_f32_16x16x32_bf16(false, pf.v,
                    false, vf.v, (short)0, o_acc[t], false, false);
        }
    }

    // ---- normalize + store O[s][h*D + d] (bf16) ----
    #pragma unroll
    for (int r = 0; r < 8; ++r) {
        const float inv = 1.0f / lrow[r];
        const int srow = qb + r + 8 * lh;
        #pragma unroll
        for (int t = 0; t < 8; ++t)
            O[(size_t)srow * HD + h * D + t * 16 + ln] = (bf16)(o_acc[t][r] * inv);
    }
}

// ---------------------------------------------------------------------------
// Host-side orchestration
// ---------------------------------------------------------------------------
extern "C" void kernel_launch(void* const* d_in, const int* in_sizes, int n_in,
                              void* d_out, int out_size, void* d_ws, size_t ws_size,
                              hipStream_t stream) {
    (void)in_sizes; (void)n_in; (void)out_size; (void)ws_size;

    const int*   x    = (const int*)d_in[0];
    const float* emb  = (const float*)d_in[1];
    const float* ff_w = (const float*)d_in[2];
    const float* q_w  = (const float*)d_in[3];
    const float* k_w  = (const float*)d_in[4];
    const float* v_w  = (const float*)d_in[5];
    const float* o_w  = (const float*)d_in[6];
    float* out = (float*)d_out;

    char* p = (char*)d_ws;
    auto alloc = [&](size_t bytes) {
        void* r = (void*)p;
        p += (bytes + 255) & ~(size_t)255;
        return r;
    };
    bf16* h_bf  = (bf16*)alloc((size_t)S * E * 2);
    bf16* hf_bf = (bf16*)alloc((size_t)S * F * 2);
    bf16* tmp   = (bf16*)alloc((size_t)S * HD * 2);
    bf16* qh    = (bf16*)alloc((size_t)H * S * D * 2);
    bf16* kh    = (bf16*)alloc((size_t)H * S * D * 2);
    bf16* vt    = (bf16*)alloc((size_t)H * D * S * 2);
    bf16* o_bf  = (bf16*)alloc((size_t)S * HD * 2);

    gather_embed<<<S, 256, 0, stream>>>(x, emb, h_bf);

    for (int i = 0; i < NL; ++i) {
        // hf = relu(h @ ff_w[i])          [S,E]x[E,F]
        gemm_bf16<true, false, bf16><<<dim3(F / 128, S / 128), 256, 0, stream>>>(
            h_bf, ff_w + (size_t)i * E * F, hf_bf, S, F, E);

        // q/k/v = hf @ {q,k,v}_w[i]       [S,F]x[F,HD]
        gemm_bf16<false, false, bf16><<<dim3(HD / 128, S / 128), 256, 0, stream>>>(
            hf_bf, q_w + (size_t)i * F * HD, tmp, S, HD, F);
        reorder_hsd<<<S, 256, 0, stream>>>(tmp, qh);

        gemm_bf16<false, false, bf16><<<dim3(HD / 128, S / 128), 256, 0, stream>>>(
            hf_bf, k_w + (size_t)i * F * HD, tmp, S, HD, F);
        reorder_hsd<<<S, 256, 0, stream>>>(tmp, kh);

        gemm_bf16<false, false, bf16><<<dim3(HD / 128, S / 128), 256, 0, stream>>>(
            hf_bf, v_w + (size_t)i * F * HD, tmp, S, HD, F);
        reorder_hds<<<S, 256, 0, stream>>>(tmp, vt);

        // causal attention
        flash_attn<<<dim3(S / 64, H), 128, 0, stream>>>(qh, kh, vt, o_bf);

        // h = o @ o_w[i]                  [S,HD]x[HD,E]
        gemm_bf16<false, false, bf16><<<dim3(E / 128, S / 128), 256, 0, stream>>>(
            o_bf, o_w + (size_t)i * HD * E, h_bf, S, E, HD);
    }

    // logits = h @ embedding^T            [S,E]x[V,E]^T -> fp32 out
    gemm_bf16<false, true, float><<<dim3(V / 128, S / 128), 256, 0, stream>>>(
        h_bf, emb, out, S, V, E);
}
